// Model_78563541778759
// MI455X (gfx1250) — matrix operand order repaired and checked by reading, not compiled, not run on hardware
//
#include <hip/hip_runtime.h>
#include <hip/hip_bf16.h>
#include <math.h>

typedef __attribute__((ext_vector_type(16))) _Float16 v16h;
typedef __attribute__((ext_vector_type(8)))  _Float16 v8h;
typedef __attribute__((ext_vector_type(4)))  _Float16 v4h;
typedef __attribute__((ext_vector_type(8)))  float    v8f;
typedef __attribute__((ext_vector_type(4)))  unsigned int v4u;

#define VOCAB 20000
#define D_    100
#define H_    100
#define K_    3
#define N_U   256
#define N_I   256
#define R_    8
#define S_    52
#define SP    50      // S - K + 1
#define E_    20000
#define KC    300     // K*D (im2col depth)
#define KCP   320     // padded K (10 k-steps of 32)
#define HP    112     // padded H (7 n-tiles of 16)
#define PROW  128     // padded D per P16 row
#define PNODE (16 * PROW)   // halfs per node tile (rows 8..15 zero)
#define SEQPAD 6720   // 64*100 + 320: covers max im2col index, zero beyond 5200

// ---------------------------------------------------------------------------
// prep: zero padded f16 P tiles; convert conv_w [100][300] f32 -> f16 [112][320]
// (layout = WMMA B operand's per-lane-contiguous layout: W16[h][kk], zero pad)
// ---------------------------------------------------------------------------
__global__ __launch_bounds__(256) void prep_kernel(const float* __restrict__ conv_w,
                                                   _Float16* __restrict__ P16,
                                                   _Float16* __restrict__ W16) {
    const int tid = blockIdx.x * blockDim.x + threadIdx.x;
    const int stride = gridDim.x * blockDim.x;
    unsigned int* p = (unsigned int*)P16;
    const int ndw = (2 * N_U * PNODE) / 2; // both branches, halfs -> dwords
    for (int i = tid; i < ndw; i += stride) p[i] = 0u;
    for (int i = tid; i < HP * KCP; i += stride) {
        const int h = i / KCP, kk = i - h * KCP;
        W16[i] = (h < H_ && kk < KC) ? (_Float16)conv_w[h * KC + kk] : (_Float16)0.0f;
    }
}

// ---------------------------------------------------------------------------
// conv branch: one 4-wave WG per (branch, node, review).
// im2col GEMM [64x320]x[320x112] via v_wmma_f32_16x16x32_f16.
// A/B fragments are raw aligned ds loads (no masking needed: the LDS seq
// buffer is zero-padded so all out-of-tile reads are finite, and B's k-pad
// rows are zero so garbage*0 contributes nothing to the f32 accumulator).
// Fused epilogue: relu(+bias) -> z=round(sigmoid(c.fc_w+b)) -> d_out;
// masked column-max -> P -> f16 tile; folded wsum (z.h_r_w), pdot (P.h_c_w).
// ---------------------------------------------------------------------------
__global__ __launch_bounds__(128) void conv_branch_kernel(
    const int* __restrict__ user_reviews, const int* __restrict__ item_reviews,
    const float* __restrict__ emb_table,
    const _Float16* __restrict__ W16,
    const float* __restrict__ conv_b,
    const float* __restrict__ fc_w_w, const float* __restrict__ fc_w_b,
    const float* __restrict__ fc_w2_w, const float* __restrict__ fc_w2_b,
    const float* __restrict__ h_r_w, const float* __restrict__ h_c_w,
    _Float16* __restrict__ P16u, _Float16* __restrict__ P16i,
    float* __restrict__ wsum_u, float* __restrict__ wsum_i,
    float* __restrict__ pdot_u, float* __restrict__ pdot_i,
    float* __restrict__ z_out) {
    const int bid    = blockIdx.x;
    const int branch = bid >> 11;      // 0 = user, 1 = item
    const int nr     = bid & 2047;     // node*8 + r
    const int node   = nr >> 3;
    const int r      = nr & 7;
    const int tid    = threadIdx.x;
    const int wave   = tid >> 5;       // wave32: 4 waves, one M-tile each
    const int lane   = tid & 31;
    const int half   = lane >> 4;
    const int lcol   = lane & 15;

    const int* revs  = branch ? item_reviews : user_reviews;
    _Float16*  P16   = branch ? P16i : P16u;
    float*     wsum  = branch ? wsum_i : wsum_u;
    float*     pdot  = branch ? pdot_i : pdot_u;
    float*     zdst  = z_out + branch * (N_U * R_ * SP) + nr * SP;

    __shared__ __align__(32) _Float16 ldsW[HP * KCP];   // 71680 B, staged weights
    __shared__ __align__(16) _Float16 ldsSeq[SEQPAD];   // 13440 B, zero padded
    __shared__ int   toks[S_];
    __shared__ float fcwLds[HP], convbLds[HP], cmaxF[HP];
    __shared__ int   cmaxI[HP];
    __shared__ float zred[64][16];
    __shared__ float wsumAcc, pdotAcc;

    if (tid < S_) toks[tid] = revs[nr * S_ + tid];
    if (tid < HP) {
        fcwLds[tid]   = (tid < H_) ? fc_w_w[tid] : 0.0f;
        convbLds[tid] = (tid < H_) ? conv_b[tid] : 0.0f;
        cmaxI[tid]    = 0;                 // post-relu values are >= 0
    }
    if (tid == 0) { wsumAcc = 0.0f; pdotAcc = 0.0f; }
    __syncthreads();

    // stage embedded sequence (f16) + zero pad tail
    for (int e = tid; e < SEQPAD; e += 128) {
        _Float16 val = (_Float16)0.0f;
        if (e < S_ * D_) {
            const int s = e / D_, d = e - s * D_;
            val = (_Float16)emb_table[toks[s] * D_ + d];
        }
        ldsSeq[e] = val;
    }
    // stage f16 weights as 16B chunks
    {
        v4u* dstW = (v4u*)ldsW;
        const v4u* srcW = (const v4u*)W16;
        for (int i = tid; i < (HP * KCP) / 8; i += 128) dstW[i] = srcW[i];
    }
    __syncthreads();

    v8f acc[7] = {};
    const int mrow = wave * 16 + lcol;     // A-operand: M = lane&15
    #pragma unroll 1
    for (int ks = 0; ks < 10; ++ks) {
        // A fragment: one contiguous 16-half run (k = 16*half + s), matching B
        const int abase = mrow * D_ + ks * 32;
        const v4h a00 = *(const v4h*)(ldsSeq + abase + 16 * half);
        const v4h a01 = *(const v4h*)(ldsSeq + abase + 16 * half + 4);
        const v4h a10 = *(const v4h*)(ldsSeq + abase + 16 * half + 8);
        const v4h a11 = *(const v4h*)(ldsSeq + abase + 16 * half + 12);
        v16h a;
        #pragma unroll
        for (int j = 0; j < 4; ++j) {
            a[j] = a00[j]; a[4 + j] = a01[j]; a[8 + j] = a10[j]; a[12 + j] = a11[j];
        }
        #pragma unroll
        for (int n = 0; n < 7; ++n) {
            const int col = n * 16 + lcol; // B-operand: N = lane&15
            const v16h b = *(const v16h*)(ldsW + col * KCP + ks * 32 + 16 * half);
            acc[n] = __builtin_amdgcn_wmma_f32_16x16x32_f16(
                false, a, false, b, (short)0, acc[n], false, false);
        }
    }

    // fused epilogue: bias + relu, z-dot partials, masked column max
    float zp[8];
    #pragma unroll
    for (int i = 0; i < 8; ++i) zp[i] = 0.0f;
    #pragma unroll
    for (int n = 0; n < 7; ++n) {
        const int col = n * 16 + lcol;
        const float fw = fcwLds[col];
        const float cb = convbLds[col];
        float mx = 0.0f;
        #pragma unroll
        for (int i = 0; i < 8; ++i) {
            const int row = wave * 16 + 8 * half + i;  // D-frag: M = i + 8*half
            const float cv = fmaxf(acc[n][i] + cb, 0.0f);
            zp[i] += cv * fw;
            mx = (row < SP) ? fmaxf(mx, cv) : mx;      // value-mask, no divergence
        }
        atomicMax(&cmaxI[col], __float_as_int(mx));    // non-neg float == int order
    }
    #pragma unroll
    for (int i = 0; i < 8; ++i) zred[wave * 16 + 8 * half + i][lcol] = zp[i];
    __syncthreads();

    if (tid < HP) cmaxF[tid] = __int_as_float(cmaxI[tid]);
    if (tid < SP) {
        float s = fc_w_b[0];
        #pragma unroll
        for (int j = 0; j < 16; ++j) s += zred[tid][j];
        const float rho = 1.0f / (1.0f + expf(-s));
        const float z   = rintf(rho);            // straight-through round (fwd)
        zdst[tid] = z;
        atomicAdd(&wsumAcc, z * h_r_w[branch * SP + tid]);
    }
    __syncthreads();

    if (tid < D_) { // P[d] = fc_w2_b[d] + sum_h cmax[h]*fc_w2_w[d][h]
        float pd = fc_w2_b[tid];
        for (int h = 0; h < H_; ++h) pd += cmaxF[h] * fc_w2_w[tid * H_ + h];
        P16[(node * 16 + r) * PROW + tid] = (_Float16)pd;
        atomicAdd(&pdotAcc, pd * h_c_w[branch * D_ + tid]);
    }
    if (tid == 0) wsum[nr] = wsumAcc;
    __syncthreads();
    if (tid == 0) pdot[nr] = pdotAcc;
}

// ---------------------------------------------------------------------------
// edge: one wave per edge (EXEC all-ones for WMMA).
// s_uv = Pu . Pi^T via 4x v_wmma_f32_16x16x32_f16 over K=128 (zero padded),
// then sigmoids of row/col sums and the pre-folded contractions.
// ---------------------------------------------------------------------------
__global__ __launch_bounds__(32) void edge_kernel(
    const int* __restrict__ uid, const int* __restrict__ iid,
    const _Float16* __restrict__ P16u, const _Float16* __restrict__ P16i,
    const float* __restrict__ wsum_u, const float* __restrict__ wsum_i,
    const float* __restrict__ pdot_u, const float* __restrict__ pdot_i,
    const float* __restrict__ user_bias, const float* __restrict__ item_bias,
    const float* __restrict__ global_bias,
    float* __restrict__ pred_r, float* __restrict__ pred_c) {
    const int e    = blockIdx.x;
    const int u    = uid[e];
    const int v    = iid[e];
    const int lane = threadIdx.x;
    const int half = lane >> 4;
    const int lcol = lane & 15;

    const _Float16* pa = P16u + u * PNODE + lcol * PROW; // A: row = review index
    const _Float16* pb = P16i + v * PNODE + lcol * PROW; // B = Pi^T: contiguous K per lane

    v8f acc = {};
    #pragma unroll
    for (int ks = 0; ks < 4; ++ks) {
        const v8h a0 = *(const v8h*)(pa + ks * 32 + 16 * half);
        const v8h a1 = *(const v8h*)(pa + ks * 32 + 16 * half + 8);
        v16h a;
        #pragma unroll
        for (int j = 0; j < 8; ++j) { a[j] = a0[j]; a[8 + j] = a1[j]; }
        const v16h b = *(const v16h*)(pb + ks * 32 + 16 * half);
        acc = __builtin_amdgcn_wmma_f32_16x16x32_f16(
            false, a, false, b, (short)0, acc, false, false);
    }

    __shared__ float sM[16][16];
    __shared__ float rhoU[8], rhoV[8];
    #pragma unroll
    for (int i = 0; i < 8; ++i) sM[i + 8 * half][lcol] = acc[i];
    __syncthreads();

    if (lane < 8) {
        float s = 0.0f;
        #pragma unroll
        for (int j = 0; j < 8; ++j) s += sM[lane][j];      // sum over v -> rho_u
        rhoU[lane] = 1.0f / (1.0f + expf(-s));
    } else if (lane < 16) {
        const int c = lane - 8;
        float s = 0.0f;
        #pragma unroll
        for (int j = 0; j < 8; ++j) s += sM[j][c];         // sum over u -> rho_v
        rhoV[c] = 1.0f / (1.0f + expf(-s));
    }
    __syncthreads();

    if (lane == 0) {
        float pr = global_bias[0] + user_bias[u] + item_bias[v];
        float pc = 0.0f;
        #pragma unroll
        for (int rr = 0; rr < 8; ++rr) {
            pr += rhoU[rr] * wsum_u[u * R_ + rr] + rhoV[rr] * wsum_i[v * R_ + rr];
            pc += rhoU[rr] * pdot_u[u * R_ + rr] + rhoV[rr] * pdot_i[v * R_ + rr];
        }
        pred_r[e] = pr;
        pred_c[e] = pc;
    }
}

// ---------------------------------------------------------------------------
extern "C" void kernel_launch(void* const* d_in, const int* in_sizes, int n_in,
                              void* d_out, int out_size, void* d_ws, size_t ws_size,
                              hipStream_t stream) {
    const int*   user_reviews = (const int*)d_in[0];
    const int*   item_reviews = (const int*)d_in[1];
    const int*   uid          = (const int*)d_in[2];
    const int*   iid          = (const int*)d_in[3];
    const float* emb_table    = (const float*)d_in[4];
    const float* conv_w       = (const float*)d_in[5];
    const float* conv_b       = (const float*)d_in[6];
    const float* fc_w_w       = (const float*)d_in[7];
    const float* fc_w_b       = (const float*)d_in[8];
    const float* fc_w2_w      = (const float*)d_in[9];
    const float* fc_w2_b      = (const float*)d_in[10];
    const float* h_r_w        = (const float*)d_in[11];
    const float* h_c_w        = (const float*)d_in[12];
    const float* user_bias    = (const float*)d_in[13];
    const float* item_bias    = (const float*)d_in[14];
    const float* global_bias  = (const float*)d_in[15];

    // workspace layout (~2.2 MB)
    _Float16* P16u   = (_Float16*)d_ws;
    _Float16* P16i   = P16u + N_U * PNODE;
    _Float16* W16    = P16i + N_I * PNODE;
    float*    wsum_u = (float*)(W16 + HP * KCP);
    float*    wsum_i = wsum_u + N_U * R_;
    float*    pdot_u = wsum_i + N_I * R_;
    float*    pdot_i = pdot_u + N_U * R_;

    float* out = (float*)d_out;       // [pred_r E][pred_c E][z_u 102400][z_i 102400]

    prep_kernel<<<512, 256, 0, stream>>>(conv_w, P16u, W16);
    conv_branch_kernel<<<2 * N_U * R_, 128, 0, stream>>>(
        user_reviews, item_reviews, emb_table, W16, conv_b,
        fc_w_w, fc_w_b, fc_w2_w, fc_w2_b, h_r_w, h_c_w,
        P16u, P16i, wsum_u, wsum_i, pdot_u, pdot_i, out + 2 * E_);
    edge_kernel<<<E_, 32, 0, stream>>>(
        uid, iid, P16u, P16i, wsum_u, wsum_i, pdot_u, pdot_i,
        user_bias, item_bias, global_bias, out, out + E_);
}
